// AdaAttN_42348377539044
// MI455X (gfx1250) — matrix operand order repaired and checked by reading, not compiled, not run on hardware
//
#include <hip/hip_runtime.h>
#include <cstdint>

typedef __attribute__((ext_vector_type(16))) __bf16 v16bf;
typedef __attribute__((ext_vector_type(8)))  __bf16 v8bf;
typedef __attribute__((ext_vector_type(4)))  __bf16 v4bf;
typedef __attribute__((ext_vector_type(8)))  float  v8f;

constexpr int B  = 4;
constexpr int C  = 512;
constexpr int L  = 4096;   // 64*64 spatial, same for content & style
constexpr int TQ = 32;     // query tile
constexpr int TK = 64;     // key tile

// ------------------------------------------------ async LDS copy (gfx1250) --
#if __has_builtin(__builtin_amdgcn_global_load_async_to_lds_b128)
#define HAVE_ASYNC_LDS 1
// Builtin expects: (int4 as(1)* src, int4 as(3)* dst, imm offset, imm cpol)
typedef int i4v __attribute__((vector_size(16)));
typedef __attribute__((address_space(1))) i4v* g128_t;
typedef __attribute__((address_space(3))) i4v* l128_t;
#else
#define HAVE_ASYNC_LDS 0
#endif

// 16B global -> LDS copy. Async (ASYNCcnt-tracked, bypasses VGPRs) when the
// gfx1250 builtin is available, else a plain vector copy.
__device__ __forceinline__ void cp_lds16(const __bf16* g, __bf16* l) {
#if HAVE_ASYNC_LDS
  __builtin_amdgcn_global_load_async_to_lds_b128(
      (g128_t)(uintptr_t)g, (l128_t)(unsigned)(uintptr_t)l, 0, 0);
#else
  *(v8bf*)l = *(const v8bf*)g;
#endif
}

template <int N>
__device__ __forceinline__ void wait_async() {
#if HAVE_ASYNC_LDS
#if __has_builtin(__builtin_amdgcn_s_wait_asynccnt)
  __builtin_amdgcn_s_wait_asynccnt(N);
#else
  asm volatile("s_wait_asynccnt %0" ::"i"(N) : "memory");
#endif
#endif
}

// ---------------------------------------------------------------- helpers ----
__device__ __forceinline__ v8f vzero8() {
  v8f z;
#pragma unroll
  for (int i = 0; i < 8; ++i) z[i] = 0.0f;
  return z;
}

// A-matrix 16x32 bf16 fragment (ISA 7.12.2): lane holds row m=lane&15;
// element e<8 -> k = half*8+e ; e>=8 -> k = 16 + half*8 + (e-8).
__device__ __forceinline__ v16bf ld_frag_a(const __bf16* row, int half) {
  v8bf lo = *(const v8bf*)(row + half * 8);
  v8bf hi = *(const v8bf*)(row + 16 + half * 8);
  return __builtin_shufflevector(lo, hi, 0, 1, 2, 3, 4, 5, 6, 7,
                                 8, 9, 10, 11, 12, 13, 14, 15);
}

// B-matrix 32x16 bf16 fragment: lane holds col n=lane&15;
// element e<8 -> k = half*8+e ; e>=8 -> k = 16 + half*8 + (e-8).
__device__ __forceinline__ v16bf ld_frag_b(const __bf16* row, int half) {
  v8bf lo = *(const v8bf*)(row + half * 8);
  v8bf hi = *(const v8bf*)(row + 16 + half * 8);
  return __builtin_shufflevector(lo, hi, 0, 1, 2, 3, 4, 5, 6, 7,
                                 8, 9, 10, 11, 12, 13, 14, 15);
}

__device__ __forceinline__ v8f wmma_bf16(v16bf a, v16bf b, v8f c) {
  return __builtin_amdgcn_wmma_f32_16x16x32_bf16(false, a, false, b,
                                                 (short)0, c, false, false);
}

// ------------------------------------------------------------- mvn stats ----
__global__ void __launch_bounds__(256) stats_kernel(const float* __restrict__ X,
                                                    float* __restrict__ mean,
                                                    float* __restrict__ istd) {
  const int c = blockIdx.x, b = blockIdx.y, tid = threadIdx.x;
  const float* p = X + ((size_t)b * C + c) * L;
  float s = 0.0f, s2 = 0.0f;
#pragma unroll 4
  for (int i = tid; i < L; i += 256) { float v = p[i]; s += v; s2 += v * v; }
  __shared__ float rs[256], rq[256];
  rs[tid] = s; rq[tid] = s2;
  __syncthreads();
  for (int off = 128; off; off >>= 1) {
    if (tid < off) { rs[tid] += rs[tid + off]; rq[tid] += rq[tid + off]; }
    __syncthreads();
  }
  if (tid == 0) {
    float m   = rs[0] / (float)L;
    float var = (rq[0] - rs[0] * rs[0] / (float)L) / (float)(L - 1);
    mean[b * C + c] = m;
    istd[b * C + c] = rsqrtf(var + 1e-5f);
  }
}

// ------------------------------------------------- 1x1-conv GEMM via WMMA ----
__global__ void __launch_bounds__(256) proj_wmma_kernel(
    const float* __restrict__ W, const float* __restrict__ bias,
    const float* __restrict__ X,
    const float* __restrict__ xmean, const float* __restrict__ xistd,
    const float* __restrict__ gamma, const float* __restrict__ beta,
    __bf16* __restrict__ Y, int transpose) {
  constexpr int TS = 40;             // 80B row stride (16B multiple)
  __shared__ __bf16 Ws[64][TS];      // A tile: [out-ch][k]       (k-contiguous)
  __shared__ __bf16 Xs[64][TS];      // B tile: [l][k] transposed (k-contiguous)

  const int b  = blockIdx.z;
  const int o0 = blockIdx.y * 64;
  const int l0 = blockIdx.x * 64;
  const int tid  = threadIdx.x;
  const int wave = tid >> 5, lane = tid & 31, half = lane >> 4, l16 = lane & 15;
  const int mt  = wave & 3;
  const int ntb = (wave >> 2) * 32;

  const float* Xb = X + (size_t)b * C * L;
  v8f accv[2];
  accv[0] = vzero8(); accv[1] = vzero8();

  for (int kb = 0; kb < C; kb += 32) {
#pragma unroll
    for (int it = 0; it < 2; ++it) {
      const int idx = it * 256 + tid;          // 0..511
      const int m = idx >> 3, kq = idx & 7;
      const float4 w4 = *(const float4*)&W[(size_t)(o0 + m) * C + kb + kq * 4];
      v4bf p;
      p[0] = (__bf16)w4.x; p[1] = (__bf16)w4.y;
      p[2] = (__bf16)w4.z; p[3] = (__bf16)w4.w;
      *(v4bf*)&Ws[m][kq * 4] = p;
    }
#pragma unroll
    for (int it = 0; it < 2; ++it) {
      const int idx = it * 256 + tid;          // 0..511
      const int k = idx >> 4, lq = idx & 15;
      const float4 x4 = *(const float4*)&Xb[(size_t)(kb + k) * L + l0 + lq * 4];
      float mu = 0.0f, is = 1.0f;
      if (xmean) { mu = xmean[b * C + kb + k]; is = xistd[b * C + kb + k]; }
      Xs[lq * 4 + 0][k] = (__bf16)((x4.x - mu) * is);
      Xs[lq * 4 + 1][k] = (__bf16)((x4.y - mu) * is);
      Xs[lq * 4 + 2][k] = (__bf16)((x4.z - mu) * is);
      Xs[lq * 4 + 3][k] = (__bf16)((x4.w - mu) * is);
    }
    __syncthreads();
    v16bf a  = ld_frag_a(&Ws[mt * 16 + l16][0], half);
    v16bf b0 = ld_frag_b(&Xs[ntb + l16][0], half);
    v16bf b1 = ld_frag_b(&Xs[ntb + 16 + l16][0], half);
    accv[0] = wmma_bf16(a, b0, accv[0]);
    accv[1] = wmma_bf16(a, b1, accv[1]);
    __syncthreads();
  }

  const int obase = o0 + mt * 16 + half * 8;   // D layout: m = r + half*8
#pragma unroll
  for (int nt = 0; nt < 2; ++nt) {
    const int l = l0 + ntb + nt * 16 + l16;    // D layout: n = lane&15
    float vr[8];
#pragma unroll
    for (int r = 0; r < 8; ++r) {
      const int o = obase + r;
      float v = accv[nt][r] + bias[o];
      if (gamma) v = v * (1.0f + gamma[b * C + o]) + beta[b * C + o];
      vr[r] = v;
    }
    if (transpose) {                           // 8 consecutive o -> one 16B store
      v8bf pk;
#pragma unroll
      for (int r = 0; r < 8; ++r) pk[r] = (__bf16)vr[r];
      *(v8bf*)&Y[(size_t)b * L * C + (size_t)l * C + obase] = pk;
    } else {
#pragma unroll
      for (int r = 0; r < 8; ++r)
        Y[(size_t)b * C * L + (size_t)(obase + r) * L + l] = (__bf16)vr[r];
    }
  }
}

// ------------------------------------------------ key_pool GEMV (1 x C) -----
__global__ void __launch_bounds__(256) keypool_kernel(
    const float* __restrict__ style, const float* __restrict__ vsp_w,
    const float* __restrict__ vsp_b, const float* __restrict__ smean,
    const float* __restrict__ sistd, float* __restrict__ kp) {
  const int b = blockIdx.y;
  const int l = blockIdx.x * 256 + threadIdx.x;
  const float* sb = style + (size_t)b * C * L;
  float acc = vsp_b[0];
#pragma unroll 4
  for (int c = 0; c < C; ++c)
    acc += vsp_w[c] * (sb[(size_t)c * L + l] - smean[b * C + c]) * sistd[b * C + c];
  kp[b * L + l] = acc;
}

// ------------------------------------------------- softmax over L, grid(B) --
__global__ void __launch_bounds__(256) softmax_kernel(float* __restrict__ kp) {
  const int b = blockIdx.x, tid = threadIdx.x;
  float* p = kp + (size_t)b * L;
  __shared__ float red[256];
  float mx = -3.0e38f;
  for (int i = tid; i < L; i += 256) mx = fmaxf(mx, p[i]);
  red[tid] = mx; __syncthreads();
  for (int off = 128; off; off >>= 1) {
    if (tid < off) red[tid] = fmaxf(red[tid], red[tid + off]);
    __syncthreads();
  }
  mx = red[0]; __syncthreads();
  float s = 0.0f;
  for (int i = tid; i < L; i += 256) s += __expf(p[i] - mx);
  red[tid] = s; __syncthreads();
  for (int off = 128; off; off >>= 1) {
    if (tid < off) red[tid] += red[tid + off];
    __syncthreads();
  }
  const float inv = 1.0f / red[0];
  for (int i = tid; i < L; i += 256) p[i] = __expf(p[i] - mx) * inv;
}

// --------------- gsv[b,c] = sum_l V[b,c,l] * sw[b,l]; grid(C,B), coalesced --
__global__ void __launch_bounds__(256) gsv_kernel(const __bf16* __restrict__ V,
                                                  const float* __restrict__ sw,
                                                  float* __restrict__ gsv) {
  const int c = blockIdx.x, b = blockIdx.y, tid = threadIdx.x;
  const __bf16* v = V + ((size_t)b * C + c) * L;
  const float*  w = sw + (size_t)b * L;
  float acc = 0.0f;
#pragma unroll 4
  for (int l = tid; l < L; l += 256) acc += (float)v[l] * w[l];
  __shared__ float red[256];
  red[tid] = acc; __syncthreads();
  for (int off = 128; off; off >>= 1) {
    if (tid < off) red[tid] += red[tid + off];
    __syncthreads();
  }
  if (tid == 0) gsv[b * C + c] = red[0];
}

// ------------------------- tiny MLP stage: out[b,o]=act(W@in+b); grid(C,B) --
__global__ void __launch_bounds__(64) mlp_kernel(const float* __restrict__ W,
                                                 const float* __restrict__ bias,
                                                 const float* __restrict__ in,
                                                 float* __restrict__ out, int relu) {
  const int o = blockIdx.x, b = blockIdx.y, tid = threadIdx.x;
  const float* w = W + (size_t)o * C;
  const float* x = in + (size_t)b * C;
  float acc = 0.0f;
  for (int k = tid; k < C; k += 64) acc += w[k] * x[k];
  __shared__ float red[64];
  red[tid] = acc; __syncthreads();
  for (int off = 32; off; off >>= 1) {
    if (tid < off) red[tid] += red[tid + off];
    __syncthreads();
  }
  if (tid == 0) {
    float v = red[0] + bias[o];
    out[b * C + o] = relu ? fmaxf(v, 0.0f) : v;
  }
}

// ------------------------------------------------- flash attention core -----
// Qt: (L x C) bf16, Kt: (L x C) bf16 (key-major), V: (C x L) bf16.
// Async-pipelined: while softmax runs, V(kt) and K(kt+1) stream into separate
// LDS buffers via GLOBAL_LOAD_ASYNC_TO_LDS_B128 (ASYNCcnt-tracked, in-order).
__global__ void __launch_bounds__(256) flash_attn_kernel(
    const __bf16* __restrict__ Qt, const __bf16* __restrict__ Kt,
    const __bf16* __restrict__ V, const float* __restrict__ content,
    const float* __restrict__ c_mean, const float* __restrict__ c_istd,
    float* __restrict__ out) {
  constexpr int QS_STRIDE  = C + 8;    // 520 elems (16B-multiple stride)
  constexpr int KST_STRIDE = C + 8;    // 520  (Kst[key][c])
  constexpr int VST_STRIDE = TK + 8;   // 72   (Vst[c][key])
  constexpr int ES_STRIDE  = TK + 4;   // 68
  constexpr size_t QS_BYTES  = (size_t)TQ * QS_STRIDE * 2;    // 33280
  constexpr size_t KST_BYTES = (size_t)TK * KST_STRIDE * 2;   // 66560
  constexpr size_t VST_BYTES = (size_t)C * VST_STRIDE * 2;    // 73728
  constexpr int NT = L / TK;

  extern __shared__ char smem[];
  __bf16* Qs  = (__bf16*)smem;                                     // [TQ][520]
  __bf16* Kst = (__bf16*)(smem + QS_BYTES);                        // [TK][520]
  __bf16* Vst = (__bf16*)(smem + QS_BYTES + KST_BYTES);            // [C][72]
  float*  Es  = (float*)(smem + QS_BYTES + KST_BYTES + VST_BYTES); // [TQ][68]
  float*  m_i  = Es + TQ * ES_STRIDE;
  float*  l_i  = m_i + TQ;
  float*  srow = l_i + TQ;

  const int b    = blockIdx.y;
  const int q0   = blockIdx.x * TQ;
  const int tid  = threadIdx.x;
  const int wave = tid >> 5, lane = tid & 31, half = lane >> 4, l16 = lane & 15;
  const int mt1 = wave & 1;        // phase-1 E tile row
  const int nt1 = wave >> 1;       // phase-1 E tile col
  const int cbase = wave * 64;     // phase-2 channel slice

  const __bf16* Qtb = Qt + (size_t)b * L * C;
  const __bf16* Ktb = Kt + (size_t)b * L * C;
  const __bf16* Vb  = V  + (size_t)b * C * L;

  // prologue: Q tile (8 chunks/thread) + K tile 0 (16 chunks/thread)
#pragma unroll
  for (int it = 0; it < 8; ++it) {
    const int idx = it * 256 + tid;
    const int q = idx >> 6, c8 = idx & 63;
    cp_lds16(&Qtb[(size_t)(q0 + q) * C + c8 * 8], &Qs[q * QS_STRIDE + c8 * 8]);
  }
#pragma unroll
  for (int it = 0; it < 16; ++it) {
    const int idx = it * 256 + tid;
    const int key = idx >> 6, c8 = idx & 63;
    cp_lds16(&Ktb[(size_t)key * C + c8 * 8], &Kst[key * KST_STRIDE + c8 * 8]);
  }
  if (tid < TQ) { m_i[tid] = -3.0e38f; l_i[tid] = 0.0f; }

  v8f acc[2][4];
#pragma unroll
  for (int i = 0; i < 2; ++i)
#pragma unroll
    for (int j = 0; j < 4; ++j) acc[i][j] = vzero8();

  wait_async<0>();
  __syncthreads();

  for (int kt = 0; kt < NT; ++kt) {
    const int k0 = kt * TK;

    // ---- phase 1: one 16x16 E tile per wave, K-loop over C=512 ----
    v8f ev = vzero8();
    const __bf16* qrow = Qs  + (mt1 * 16 + l16) * QS_STRIDE;
    const __bf16* krow = Kst + (nt1 * 16 + l16) * KST_STRIDE;
#pragma unroll
    for (int kk = 0; kk < C; kk += 32)
      ev = wmma_bf16(ld_frag_a(qrow + kk, half), ld_frag_b(krow + kk, half), ev);
#pragma unroll
    for (int r = 0; r < 8; ++r)
      Es[(mt1 * 16 + half * 8 + r) * ES_STRIDE + nt1 * 16 + l16] = ev[r];
    __syncthreads();   // E complete; all waves done reading Kst

    // ---- prefetch: V(kt) then K(kt+1) (16 async instrs each, in order) ----
#pragma unroll
    for (int it = 0; it < 16; ++it) {
      const int idx = it * 256 + tid;
      const int c = idx >> 3, k8 = idx & 7;
      cp_lds16(&Vb[(size_t)c * L + k0 + k8 * 8], &Vst[c * VST_STRIDE + k8 * 8]);
    }
    if (kt + 1 < NT) {
#pragma unroll
      for (int it = 0; it < 16; ++it) {
        const int idx = it * 256 + tid;
        const int key = idx >> 6, c8 = idx & 63;
        cp_lds16(&Ktb[(size_t)(k0 + TK + key) * C + c8 * 8],
                 &Kst[key * KST_STRIDE + c8 * 8]);
      }
    }

    // ---- online softmax update (overlaps the async DMA) ----
    if (tid < TQ) {
      float mold = m_i[tid];
      float mx = mold;
      const int base = tid * ES_STRIDE;
      for (int j = 0; j < TK; ++j) mx = fmaxf(mx, Es[base + j]);
      float sc = __expf(mold - mx);
      float s = 0.0f;
      for (int j = 0; j < TK; ++j) {
        float p = __expf(Es[base + j] - mx);
        Es[base + j] = p;
        s += p;
      }
      l_i[tid]  = l_i[tid] * sc + s;
      m_i[tid]  = mx;
      srow[tid] = sc;
    }
    if (kt + 1 < NT) wait_async<16>();  // first 16 (V) done; K still in flight
    else             wait_async<0>();
    __syncthreads();   // everyone's V tile + softmax visible

    // ---- rescale accumulators ----
#pragma unroll
    for (int mt = 0; mt < 2; ++mt) {
      float scm[8];
#pragma unroll
      for (int r = 0; r < 8; ++r) scm[r] = srow[mt * 16 + half * 8 + r];
#pragma unroll
      for (int nt = 0; nt < 4; ++nt)
#pragma unroll
        for (int r = 0; r < 8; ++r) acc[mt][nt][r] *= scm[r];
    }

    // ---- phase 2: acc += S(32x64) * V^T(64 x 512-slice) ----
#pragma unroll
    for (int ks = 0; ks < 2; ++ks) {
      v16bf af[2];
#pragma unroll
      for (int mt = 0; mt < 2; ++mt) {
        const float* erow = Es + (mt * 16 + l16) * ES_STRIDE + ks * 32;
        v16bf a;
#pragma unroll
        for (int e = 0; e < 8; ++e) a[e] = (__bf16)erow[half * 8 + e];
#pragma unroll
        for (int e = 0; e < 8; ++e) a[8 + e] = (__bf16)erow[16 + half * 8 + e];
        af[mt] = a;
      }
#pragma unroll
      for (int nt = 0; nt < 4; ++nt) {
        const __bf16* vrow = Vst + (cbase + nt * 16 + l16) * VST_STRIDE + ks * 32;
        v16bf bf = ld_frag_b(vrow, half);
#pragma unroll
        for (int mt = 0; mt < 2; ++mt)
          acc[mt][nt] = wmma_bf16(af[mt], bf, acc[mt][nt]);
      }
    }
    wait_async<0>();   // next K tile fully in LDS
    __syncthreads();   // all waves done with Vst/Es; Kst ready everywhere
  }

  // ---- epilogue: out[b,c,q] = acc/l_i + (content - mean)*istd, float4 IO ----
  const float* cm   = c_mean + b * C;
  const float* ci   = c_istd + b * C;
  const float* cont = content + (size_t)b * C * L;
  float*       ob   = out + (size_t)b * C * L;
#pragma unroll
  for (int mt = 0; mt < 2; ++mt) {
    const int qb = q0 + mt * 16 + half * 8;
    float inv[8];
#pragma unroll
    for (int r = 0; r < 8; ++r) inv[r] = 1.0f / l_i[mt * 16 + half * 8 + r];
#pragma unroll
    for (int nt = 0; nt < 4; ++nt) {
      const int ch = cbase + nt * 16 + l16;
      const float mean = cm[ch], istd = ci[ch];
      const size_t base = (size_t)ch * L + qb;
      const float4 c0 = *(const float4*)&cont[base];
      const float4 c1 = *(const float4*)&cont[base + 4];
      float4 o0, o1;
      o0.x = acc[mt][nt][0] * inv[0] + (c0.x - mean) * istd;
      o0.y = acc[mt][nt][1] * inv[1] + (c0.y - mean) * istd;
      o0.z = acc[mt][nt][2] * inv[2] + (c0.z - mean) * istd;
      o0.w = acc[mt][nt][3] * inv[3] + (c0.w - mean) * istd;
      o1.x = acc[mt][nt][4] * inv[4] + (c1.x - mean) * istd;
      o1.y = acc[mt][nt][5] * inv[5] + (c1.y - mean) * istd;
      o1.z = acc[mt][nt][6] * inv[6] + (c1.z - mean) * istd;
      o1.w = acc[mt][nt][7] * inv[7] + (c1.w - mean) * istd;
      *(float4*)&ob[base]     = o0;
      *(float4*)&ob[base + 4] = o1;
    }
  }
}

// ---------------------------------------------------------------- driver ----
extern "C" void kernel_launch(void* const* d_in, const int* in_sizes, int n_in,
                              void* d_out, int out_size, void* d_ws, size_t ws_size,
                              hipStream_t stream) {
  (void)in_sizes; (void)n_in; (void)out_size; (void)ws_size;
  const float* content = (const float*)d_in[0];
  const float* style   = (const float*)d_in[1];
  const float* v_w   = (const float*)d_in[2];
  const float* v_b   = (const float*)d_in[3];
  const float* vsp_w = (const float*)d_in[4];
  const float* vsp_b = (const float*)d_in[5];
  const float* k_w   = (const float*)d_in[6];
  const float* k_b   = (const float*)d_in[7];
  const float* qg_w  = (const float*)d_in[8];
  const float* qg_b  = (const float*)d_in[9];
  const float* g1_w1 = (const float*)d_in[10];
  const float* g1_b1 = (const float*)d_in[11];
  const float* g1_w2 = (const float*)d_in[12];
  const float* g1_b2 = (const float*)d_in[13];
  const float* g2_w1 = (const float*)d_in[14];
  const float* g2_b1 = (const float*)d_in[15];
  const float* g2_w2 = (const float*)d_in[16];
  const float* g2_b2 = (const float*)d_in[17];
  float* out = (float*)d_out;

  char* ws = (char*)d_ws;
  size_t off = 0;
  auto carve = [&](size_t bytes) -> char* {
    char* p = ws + off;
    off += (bytes + 255) & ~(size_t)255;
    return p;
  };
  float* s_mean = (float*)carve((size_t)B * C * 4);
  float* s_istd = (float*)carve((size_t)B * C * 4);
  float* c_mean = (float*)carve((size_t)B * C * 4);
  float* c_istd = (float*)carve((size_t)B * C * 4);
  float* kp     = (float*)carve((size_t)B * L * 4);   // key_pool -> softmax weights
  float* gsv    = (float*)carve((size_t)B * C * 4);
  float* hbuf   = (float*)carve((size_t)B * C * 4);
  float* gammab = (float*)carve((size_t)B * C * 4);
  float* betab  = (float*)carve((size_t)B * C * 4);
  __bf16* Vb = (__bf16*)carve((size_t)B * C * L * 2); // V  (C x L) bf16
  __bf16* Kt = (__bf16*)carve((size_t)B * L * C * 2); // K^T (L x C) bf16
  __bf16* Qt = (__bf16*)carve((size_t)B * L * C * 2); // Q^T (L x C) bf16

  // 1. mvn statistics
  stats_kernel<<<dim3(C, B), 256, 0, stream>>>(style,   s_mean, s_istd);
  stats_kernel<<<dim3(C, B), 256, 0, stream>>>(content, c_mean, c_istd);

  // 2. V, K projections (WMMA). V stays (C,L); K stored key-major.
  proj_wmma_kernel<<<dim3(L / 64, C / 64, B), 256, 0, stream>>>(
      v_w, v_b, style, nullptr, nullptr, nullptr, nullptr, Vb, 0);
  proj_wmma_kernel<<<dim3(L / 64, C / 64, B), 256, 0, stream>>>(
      k_w, k_b, style, nullptr, nullptr, nullptr, nullptr, Kt, 1);

  // 3. style pooling path
  keypool_kernel<<<dim3(L / 256, B), 256, 0, stream>>>(style, vsp_w, vsp_b,
                                                       s_mean, s_istd, kp);
  softmax_kernel<<<dim3(B), 256, 0, stream>>>(kp);
  gsv_kernel<<<dim3(C, B), 256, 0, stream>>>(Vb, kp, gsv);

  // 4. gamma / beta MLPs
  mlp_kernel<<<dim3(C, B), 64, 0, stream>>>(g1_w1, g1_b1, gsv, hbuf, 1);
  mlp_kernel<<<dim3(C, B), 64, 0, stream>>>(g1_w2, g1_b2, hbuf, gammab, 0);
  mlp_kernel<<<dim3(C, B), 64, 0, stream>>>(g2_w1, g2_b1, gsv, hbuf, 1);
  mlp_kernel<<<dim3(C, B), 64, 0, stream>>>(g2_w2, g2_b2, hbuf, betab, 0);

  // 5. Q projection with fused mvn input + (1+gamma)*x+beta epilogue (WMMA)
  proj_wmma_kernel<<<dim3(L / 64, C / 64, B), 256, 0, stream>>>(
      qg_w, qg_b, content, c_mean, c_istd, gammab, betab, Qt, 1);

  // 6. flash attention + fused mvn(content) residual (async-pipelined K/V)
  constexpr size_t SMEM = (size_t)TQ * (C + 8) * 2          // Qs
                        + (size_t)TK * (C + 8) * 2          // K tile
                        + (size_t)C * (TK + 8) * 2          // V tile
                        + (size_t)TQ * (TK + 4) * 4         // scores
                        + (size_t)3 * TQ * 4;               // m_i, l_i, srow
  flash_attn_kernel<<<dim3(L / TQ, B), 256, SMEM, stream>>>(
      Qt, Kt, Vb, content, c_mean, c_istd, out);
}
